// GraphActorCriticNetwork_19954418057371
// MI455X (gfx1250) — matrix operand order repaired and checked by reading, not compiled, not run on hardware
//
#include <hip/hip_runtime.h>
#include <hip/hip_bf16.h>

typedef _Float16 v16h __attribute__((ext_vector_type(16)));
typedef _Float16 v8h  __attribute__((ext_vector_type(8)));
typedef float    v8f  __attribute__((ext_vector_type(8)));

// Dimensions (fixed by the reference)
#define NN   21
#define FF   2048
#define EE   64
#define NE   85      // E + N self loops
#define FC1N 1024
#define FC2N 512

__device__ __forceinline__ v16h cat16(v8h lo, v8h hi) {
  return __builtin_shufflevector(lo, hi, 0, 1, 2, 3, 4, 5, 6, 7,
                                         8, 9, 10, 11, 12, 13, 14, 15);
}

// ---------------------------------------------------------------------------
// Pack: row-major f32 [M,K] -> row-major f16 [Mpad,Kpad], zero padded.
// ---------------------------------------------------------------------------
__global__ __launch_bounds__(256) void k_pack_pad(const float* __restrict__ src,
                                                  _Float16* __restrict__ dst,
                                                  int M, int K, int Kpad, int total) {
  const int gid = blockIdx.x * 256 + threadIdx.x;
  if (gid >= total) return;
  const int r = gid / Kpad, k = gid - r * Kpad;
  dst[gid] = (r < M && k < K) ? (_Float16)src[r * K + k] : (_Float16)0.0f;
}

// ---------------------------------------------------------------------------
// Pack transpose: f32 [K,N] row-major -> f16 [Npad,Kpad] (dst[n][k] = src[k][n]).
// gid is k-major so the f32 reads are coalesced along N.
// ---------------------------------------------------------------------------
__global__ __launch_bounds__(256) void k_pack_tr(const float* __restrict__ src,
                                                 _Float16* __restrict__ dst,
                                                 int K, int N, int Kpad, int Npad,
                                                 int total) {
  const int gid = blockIdx.x * 256 + threadIdx.x;
  if (gid >= total) return;
  const int k = gid / Npad, n = gid - k * Npad;
  const _Float16 v = (k < K && n < N) ? (_Float16)src[k * N + n] : (_Float16)0.0f;
  dst[n * Kpad + k] = v;
}

// ---------------------------------------------------------------------------
// K1: h1p[21,21] = A1h(32x2048 padded) @ B1t(32x2048, N-major) via f16 WMMA.
// 1 block, 4 waves; wave w -> 16x16 tile (w>>1, w&1). Unconditional b128 loads.
// ---------------------------------------------------------------------------
__global__ __launch_bounds__(128) void k_gemm1(const _Float16* __restrict__ A1h,
                                               const _Float16* __restrict__ B1t,
                                               float* __restrict__ h1p) {
  const int lane = threadIdx.x & 31;
  const int wv   = threadIdx.x >> 5;
  const int ti = wv >> 1, tj = wv & 1;
  const int half = lane >> 4;
  const int l15  = lane & 15;
  const _Float16* __restrict__ arow = A1h + (ti * 16 + l15) * FF;
  const _Float16* __restrict__ brow = B1t + (tj * 16 + l15) * FF;
  v8f c = {};
  for (int kb = 0; kb < FF; kb += 32) {
    const v8h a0 = *(const v8h*)(arow + kb + 8 * half);
    const v8h a1 = *(const v8h*)(arow + kb + 16 + 8 * half);
    const v8h b0 = *(const v8h*)(brow + kb + 8 * half);
    const v8h b1 = *(const v8h*)(brow + kb + 16 + 8 * half);
    c = __builtin_amdgcn_wmma_f32_16x16x32_f16(false, cat16(a0, a1),
                                               false, cat16(b0, b1),
                                               (short)0, c, false, false);
  }
#pragma unroll
  for (int r = 0; r < 8; ++r) {             // D: M = r + 8*half, N = lane&15
    const int gm = ti * 16 + r + 8 * half;
    const int gn = tj * 16 + l15;
    if (gm < NN && gn < NN) h1p[gm * NN + gn] = c[r];
  }
}

// ---------------------------------------------------------------------------
// K2: GCN aggregation #1: degrees, edge norms, scatter-add, +bias.
// ---------------------------------------------------------------------------
__global__ __launch_bounds__(448) void k_agg1(const int* __restrict__ adj,
                                              const float* __restrict__ h1p,
                                              const float* __restrict__ b1,
                                              float* __restrict__ norm_ws,
                                              float* __restrict__ x1) {
  __shared__ float sdeg[NN];
  __shared__ float snorm[96];
  __shared__ int   ssrc[96], sdst[96];
  const int t = threadIdx.x;
  if (t < NE) {
    int s, d;
    if (t < EE) { s = adj[t]; d = adj[EE + t]; }
    else        { s = t - EE; d = t - EE; }
    ssrc[t] = s; sdst[t] = d;
  }
  __syncthreads();
  if (t < NN) {
    float deg = 0.f;
    for (int i = 0; i < NE; ++i) deg += (sdst[i] == t) ? 1.f : 0.f;
    sdeg[t] = deg;
  }
  __syncthreads();
  if (t < NE) {
    const float nv = rsqrtf(sdeg[ssrc[t]]) * rsqrtf(sdeg[sdst[t]]);
    snorm[t] = nv;
    norm_ws[t] = nv;                      // reused by K3
  }
  __syncthreads();
  if (t < NN * NN) {
    const int d = t / NN, f = t % NN;
    float acc = 0.f;
    for (int i = 0; i < NE; ++i)
      if (sdst[i] == d) acc += snorm[i] * h1p[ssrc[i] * NN + f];
    x1[t] = acc + b1[f];
  }
}

// ---------------------------------------------------------------------------
// K3: gc2 projection (21x21 @ 21x21) + aggregation + bias -> flat_h[448] f16
// ---------------------------------------------------------------------------
__global__ __launch_bounds__(448) void k_gc2(const int* __restrict__ adj,
                                             const float* __restrict__ x1,
                                             const float* __restrict__ w2,
                                             const float* __restrict__ b2,
                                             const float* __restrict__ norm_ws,
                                             _Float16* __restrict__ flat_h) {
  __shared__ float sx[NN * NN];
  __shared__ float sh[NN * NN];
  __shared__ float snorm[96];
  __shared__ int   ssrc[96], sdst[96];
  const int t = threadIdx.x;
  if (t < NN * NN) sx[t] = x1[t];
  if (t < NE) {
    int s, d;
    if (t < EE) { s = adj[t]; d = adj[EE + t]; }
    else        { s = t - EE; d = t - EE; }
    ssrc[t] = s; sdst[t] = d; snorm[t] = norm_ws[t];
  }
  __syncthreads();
  if (t < NN * NN) {
    const int nd = t / NN, f = t % NN;
    float acc = 0.f;
    for (int k = 0; k < NN; ++k) acc += sx[nd * NN + k] * w2[k * NN + f];
    sh[t] = acc;
  }
  __syncthreads();
  float val = 0.f;
  if (t < NN * NN) {
    const int d = t / NN, f = t % NN;
    float acc = 0.f;
    for (int i = 0; i < NE; ++i)
      if (sdst[i] == d) acc += snorm[i] * sh[ssrc[i] * NN + f];
    val = acc + b2[f];
  }
  if (t < 448) flat_h[t] = (_Float16)val;   // zero padded to 448 for WMMA K
}

// ---------------------------------------------------------------------------
// K4: fc1: relu(flat @ fc1_w + b) via WMMA, M padded to 16 (row 0 real).
// A loads are unconditional (padded vector) + branchless lane select.
// Wt is fc1_w^T packed [1024][448] f16. 16 blocks x 4 waves, one N tile each.
// ---------------------------------------------------------------------------
__global__ __launch_bounds__(128) void k_fc1(const _Float16* __restrict__ vec,
                                             const _Float16* __restrict__ Wt,
                                             const float* __restrict__ bias,
                                             _Float16* __restrict__ out_h) {
  const int lane = threadIdx.x & 31;
  const int wv   = threadIdx.x >> 5;
  const int n0   = (blockIdx.x * 4 + wv) * 16;
  const int half = lane >> 4;
  const int l15  = lane & 15;
  const _Float16* __restrict__ brow = Wt + (n0 + l15) * 448;
  const bool row0 = (l15 == 0);
  const v16h zero = {};
  v8f c = {};
  for (int kb = 0; kb < 448; kb += 32) {
    const v8h a0 = *(const v8h*)(vec + kb + 8 * half);
    const v8h a1 = *(const v8h*)(vec + kb + 16 + 8 * half);
    const v8h b0 = *(const v8h*)(brow + kb + 8 * half);
    const v8h b1 = *(const v8h*)(brow + kb + 16 + 8 * half);
    const v16h a = row0 ? cat16(a0, a1) : zero;
    c = __builtin_amdgcn_wmma_f32_16x16x32_f16(false, a,
                                               false, cat16(b0, b1),
                                               (short)0, c, false, false);
  }
  if (lane < 16) {                          // row M=0 -> c[0], lanes 0..15
    const float v = c[0] + bias[n0 + lane];
    out_h[n0 + lane] = (_Float16)fmaxf(v, 0.f);
  }
}

// ---------------------------------------------------------------------------
// K5: fc2: relu(h1024 @ fc2_w + b) -> f32[512]. Wt = fc2_w^T [512][1024] f16.
// ---------------------------------------------------------------------------
__global__ __launch_bounds__(128) void k_fc2(const _Float16* __restrict__ vec,
                                             const _Float16* __restrict__ Wt,
                                             const float* __restrict__ bias,
                                             float* __restrict__ out_f) {
  const int lane = threadIdx.x & 31;
  const int wv   = threadIdx.x >> 5;
  const int n0   = (blockIdx.x * 4 + wv) * 16;
  const int half = lane >> 4;
  const int l15  = lane & 15;
  const _Float16* __restrict__ brow = Wt + (n0 + l15) * FC1N;
  const bool row0 = (l15 == 0);
  const v16h zero = {};
  v8f c = {};
  for (int kb = 0; kb < FC1N; kb += 32) {
    const v8h a0 = *(const v8h*)(vec + kb + 8 * half);
    const v8h a1 = *(const v8h*)(vec + kb + 16 + 8 * half);
    const v8h b0 = *(const v8h*)(brow + kb + 8 * half);
    const v8h b1 = *(const v8h*)(brow + kb + 16 + 8 * half);
    const v16h a = row0 ? cat16(a0, a1) : zero;
    c = __builtin_amdgcn_wmma_f32_16x16x32_f16(false, a,
                                               false, cat16(b0, b1),
                                               (short)0, c, false, false);
  }
  if (lane < 16) {
    const float v = c[0] + bias[n0 + lane];
    out_f[n0 + lane] = fmaxf(v, 0.f);
  }
}

// ---------------------------------------------------------------------------
// K6: heads: fc3->fc4->v and pi, f32 VALU (tiny). out = [pi(132) | v(1)].
// ---------------------------------------------------------------------------
__global__ __launch_bounds__(256) void k_heads(const float* __restrict__ h2,
                                               const float* __restrict__ fc3_w,
                                               const float* __restrict__ fc3_b,
                                               const float* __restrict__ fc4_w,
                                               const float* __restrict__ fc4_b,
                                               const float* __restrict__ pi_w,
                                               const float* __restrict__ pi_b,
                                               const float* __restrict__ v_w,
                                               const float* __restrict__ v_b,
                                               float* __restrict__ out) {
  __shared__ float sh2[FC2N];
  __shared__ float s3[256];
  __shared__ float s4[64];
  const int t = threadIdx.x;
  sh2[t] = h2[t];
  sh2[t + 256] = h2[t + 256];
  __syncthreads();
  {
    float acc = fc3_b[t];
    for (int k = 0; k < FC2N; ++k) acc += sh2[k] * fc3_w[k * 256 + t];
    s3[t] = fmaxf(acc, 0.f);
  }
  __syncthreads();
  if (t < 64) {
    float acc = fc4_b[t];
    for (int k = 0; k < 256; ++k) acc += s3[k] * fc4_w[k * 64 + t];
    s4[t] = fmaxf(acc, 0.f);
  }
  if (t < 132) {
    float acc = pi_b[t];
    for (int k = 0; k < FC2N; ++k) acc += sh2[k] * pi_w[k * 132 + t];
    out[t] = acc;
  }
  __syncthreads();
  if (t == 0) {
    float acc = v_b[0];
    for (int k = 0; k < 64; ++k) acc += s4[k] * v_w[k];
    out[132] = acc;
  }
}

// ---------------------------------------------------------------------------
extern "C" void kernel_launch(void* const* d_in, const int* in_sizes, int n_in,
                              void* d_out, int out_size, void* d_ws, size_t ws_size,
                              hipStream_t stream) {
  const float* state = (const float*)d_in[0];
  const int*   adj   = (const int*)d_in[1];
  const float* gc1_w = (const float*)d_in[2];
  const float* gc1_b = (const float*)d_in[3];
  const float* gc2_w = (const float*)d_in[4];
  const float* gc2_b = (const float*)d_in[5];
  const float* fc1_w = (const float*)d_in[6];
  const float* fc1_b = (const float*)d_in[7];
  const float* fc2_w = (const float*)d_in[8];
  const float* fc2_b = (const float*)d_in[9];
  const float* fc3_w = (const float*)d_in[10];
  const float* fc3_b = (const float*)d_in[11];
  const float* fc4_w = (const float*)d_in[12];
  const float* fc4_b = (const float*)d_in[13];
  const float* pi_w  = (const float*)d_in[14];
  const float* pi_b  = (const float*)d_in[15];
  const float* v_w   = (const float*)d_in[16];
  const float* v_b   = (const float*)d_in[17];
  float* out = (float*)d_out;

  char* ws = (char*)d_ws;
  _Float16* A1h  = (_Float16*)(ws + 0);        // [32][2048]   131072 B
  _Float16* B1t  = (_Float16*)(ws + 131072);   // [32][2048]   131072 B
  _Float16* W1t  = (_Float16*)(ws + 262144);   // [1024][448]  917504 B
  _Float16* W2t  = (_Float16*)(ws + 1179648);  // [512][1024] 1048576 B
  float*    norm_ws = (float*)(ws + 2228224);  //  85 f32
  float*    h1p     = (float*)(ws + 2228736);  // 441 f32
  float*    x1      = (float*)(ws + 2230784);  // 441 f32
  _Float16* flat_h  = (_Float16*)(ws + 2232832); // 448 f16 (padded)
  _Float16* fc1_h   = (_Float16*)(ws + 2233856); // 1024 f16
  float*    fc2_out = (float*)(ws + 2236416);    // 512 f32

  // Stage operands as padded f16 (A row-major, weights N-major/transposed).
  {
    int tA = 32 * FF;
    k_pack_pad<<<(tA + 255) / 256, 256, 0, stream>>>(state, A1h, NN, FF, FF, tA);
    int tB = FF * 32;
    k_pack_tr<<<(tB + 255) / 256, 256, 0, stream>>>(gc1_w, B1t, FF, NN, FF, 32, tB);
    int t1 = 448 * FC1N;
    k_pack_tr<<<(t1 + 255) / 256, 256, 0, stream>>>(fc1_w, W1t, NN * NN, FC1N, 448, FC1N, t1);
    int t2 = FC1N * FC2N;
    k_pack_tr<<<(t2 + 255) / 256, 256, 0, stream>>>(fc2_w, W2t, FC1N, FC2N, FC1N, FC2N, t2);
  }

  k_gemm1<<<1, 128, 0, stream>>>(A1h, B1t, h1p);
  k_agg1 <<<1, 448, 0, stream>>>(adj, h1p, gc1_b, norm_ws, x1);
  k_gc2  <<<1, 448, 0, stream>>>(adj, x1, gc2_w, gc2_b, norm_ws, flat_h);
  k_fc1  <<<16, 128, 0, stream>>>(flat_h, W1t, fc1_b, fc1_h);
  k_fc2  <<<8, 128, 0, stream>>>(fc1_h, W2t, fc2_b, fc2_out);
  k_heads<<<1, 256, 0, stream>>>(fc2_out, fc3_w, fc3_b, fc4_w, fc4_b,
                                 pi_w, pi_b, v_w, v_b, out);
}
